// MultiScaleSNNEncoderDet_30099130810827
// MI455X (gfx1250) — matrix operand order repaired and checked by reading, not compiled, not run on hardware
//
#include <hip/hip_runtime.h>
#include <hip/hip_bf16.h>

// ---------------------------------------------------------------------------
// Types for WMMA operands (trivially-copyable, safe in unions)
// ---------------------------------------------------------------------------
typedef __attribute__((ext_vector_type(16))) __bf16 v16bf;
typedef __attribute__((ext_vector_type(8)))  float  v8f;
typedef __attribute__((ext_vector_type(4)))  float  f32x4;

union BF16x16 {
    v16bf v;
    f32x4 f4[2];
    unsigned short u[16];
};

__device__ __forceinline__ int imin(int a, int b) { return a < b ? a : b; }
__device__ __forceinline__ int imax(int a, int b) { return a > b ? a : b; }

__device__ __forceinline__ unsigned short f2bf(float f) {
    unsigned int u = __float_as_uint(f);
    unsigned int r = u + 0x7FFFu + ((u >> 16) & 1u);   // round-to-nearest-even
    return (unsigned short)(r >> 16);
}

__device__ __forceinline__ v8f wmma_bf16(const BF16x16& a, const BF16x16& b, v8f c) {
    return __builtin_amdgcn_wmma_f32_16x16x32_bf16(
        /*neg_a=*/false, a.v, /*neg_b=*/false, b.v,
        /*c_mod=*/(short)0, c, /*reuse_a=*/false, /*reuse_b=*/false);
}

// ---------------------------------------------------------------------------
// Zero a small page (used as padding source for conv3x3 B-operand loads)
// ---------------------------------------------------------------------------
__global__ void zero_kernel(unsigned short* p) { p[threadIdx.x] = 0; }

// ---------------------------------------------------------------------------
// Average pooling: out[b][c][y][x] = mean of k x k block of in (NCHW f32)
// ---------------------------------------------------------------------------
template<int KP, int Ho, int Wo>
__global__ void pool_kernel(const float* __restrict__ in, float* __restrict__ out) {
    size_t idx = (size_t)blockIdx.x * 256 + threadIdx.x;
    constexpr size_t total = (size_t)16 * 128 * Ho * Wo;
    if (idx >= total) return;
    int x  = (int)(idx % Wo);
    int y  = (int)((idx / Wo) % Ho);
    size_t bc = idx / ((size_t)Wo * Ho);
    constexpr int Wi = Wo * KP;
    const float* p = in + bc * (size_t)(Ho * KP) * Wi + (size_t)(y * KP) * Wi + (size_t)x * KP;
    float s = 0.f;
#pragma unroll
    for (int i = 0; i < KP; ++i)
#pragma unroll
        for (int j = 0; j < KP; ++j)
            s += p[i * Wi + j];
    out[idx] = s * (1.f / (KP * KP));
}

// ---------------------------------------------------------------------------
// NCHW f32 -> NHWC bf16 transpose-cast (LDS tiled; producer-side conversion
// so GEMM inner loops never convert). Tile: 128 channels x 16 pixels.
// ---------------------------------------------------------------------------
template<int HW>
__global__ void castT_kernel(const float* __restrict__ in, unsigned short* __restrict__ out) {
    __shared__ unsigned short t[16 * 132];           // pad 132 to dodge bank conflicts
    const int b = blockIdx.y, n0 = blockIdx.x * 16, tid = threadIdx.x;
    const float* src = in + (size_t)b * 128 * HW + n0;
    for (int i = tid; i < 2048; i += 256) {
        int c = i >> 4, n = i & 15;                  // n fastest: coalesced global reads
        t[n * 132 + c] = f2bf(src[c * HW + n]);
    }
    __syncthreads();
    unsigned short* dst = out + ((size_t)b * HW + n0) * 128;
    for (int i = tid; i < 2048; i += 256)            // c fastest: coalesced global writes
        dst[i] = t[(i >> 7) * 132 + (i & 127)];
}

// ---------------------------------------------------------------------------
// Fused conv1x1 GEMM (bf16 WMMA) + BatchNorm + epilogue.
//   Y[b][co][n] = act( BN( sum_ci W[co][ci] * X[b][n][ci] ) )
// X is NHWC bf16: each B-operand K-chunk = two contiguous b128 loads.
// A (weights) staged in LDS bf16, fetched as 16B ds_load runs matching the
// CDNA5 A layout  K(e,lane) = e + 8*((e>>3) + (lane>>4)).
// MODE: 0 = bf16 NHWC store (V), 1 = ternary bf16 NCHW store (Q/K),
//       2 = +resid(f32 NCHW), bf16 NHWC store (F)
// ---------------------------------------------------------------------------
template<int HW, int MODE>
__global__ void gemm_bn_wmma(const unsigned short* __restrict__ X,   // [B][HW][128] bf16
                             const float* __restrict__ Wt,
                             unsigned long long wBatchStride,
                             const float* __restrict__ bng, const float* __restrict__ bnb,
                             const float* __restrict__ bnm, const float* __restrict__ bnv,
                             const float* __restrict__ resid,        // [B][128][HW] f32
                             unsigned short* __restrict__ outT) {
    __shared__ __align__(16) unsigned short wlds[16 * 128];
    __shared__ float sc[16], bs[16];

    const int b      = blockIdx.z;
    const int cobase = blockIdx.y * 16;
    const int tid    = threadIdx.x;

    const float* wsrc = Wt + (size_t)b * wBatchStride;
    for (int i = tid; i < 16 * 128; i += 256) {
        int m = i >> 7, ci = i & 127;
        wlds[i] = f2bf(wsrc[(size_t)(cobase + m) * 128 + ci]);
    }
    if (tid < 16) {
        if (bng) {
            float inv = bng[cobase + tid] * rsqrtf(bnv[cobase + tid] + 1e-5f);
            sc[tid] = inv;
            bs[tid] = bnb[cobase + tid] - bnm[cobase + tid] * inv;
        } else {
            sc[tid] = 1.f; bs[tid] = 0.f;
        }
    }
    __syncthreads();

    const int wave = tid >> 5, lane = tid & 31;
    const int hi = lane >> 4, lm = lane & 15;
    const int ncol = blockIdx.x * 128 + wave * 16 + lm;
    const bool inb = ncol < HW;
    const int nc = imin(ncol, HW - 1);               // clamped, always in-bounds

    const unsigned short* xrow = X + ((size_t)b * HW + nc) * 128;
    v8f acc = {};

#pragma unroll
    for (int k0 = 0; k0 < 128; k0 += 32) {
        BF16x16 a, bx;
        const unsigned short* wp = &wlds[lm * 128 + k0];
        a.f4[0] = *(const f32x4*)(wp + 8 * hi);          // K = {0..7}+8*hi
        a.f4[1] = *(const f32x4*)(wp + 16 + 8 * hi);     // K = {16..23}+8*hi
        const unsigned short* xp = xrow + k0 + 8 * hi;   // K = {0..7,16..23}+8*hi
        bx.f4[0] = *(const f32x4*)(xp);
        bx.f4[1] = *(const f32x4*)(xp + 16);
        acc = wmma_bf16(a, bx, acc);
    }

    if (inb) {
        if constexpr (MODE == 1) {                    // ternary -> bf16, NCHW
#pragma unroll
            for (int r = 0; r < 8; ++r) {
                int m  = r + 8 * hi;
                float v = acc[r] * sc[m] + bs[m];
                size_t oidx = ((size_t)b * 128 + cobase + m) * HW + ncol;
                outT[oidx] = (v > 0.f) ? 0x3F80u : ((v < 0.f) ? 0xBF80u : 0u);
            }
        } else {                                      // bf16, NHWC vector store
            union { f32x4 f; unsigned short us[8]; } pk;
#pragma unroll
            for (int r = 0; r < 8; ++r) {
                int m  = r + 8 * hi;
                float v = acc[r] * sc[m] + bs[m];
                if constexpr (MODE == 2)
                    v += resid[((size_t)b * 128 + cobase + m) * HW + ncol];
                pk.us[r] = f2bf(v);
            }
            *(f32x4*)(outT + ((size_t)b * HW + ncol) * 128 + cobase + 8 * hi) = pk.f;
        }
    }
}

// ---------------------------------------------------------------------------
// Attention: S[d][e] = sum_n q[d][n]*k[e][n]  (WMMA over K = spatial N),
// softmax over e, store attn[b][h][d][e].  One workgroup per (head, batch).
// ---------------------------------------------------------------------------
template<int HW>
__global__ void attn_kernel(const unsigned short* __restrict__ Q,
                            const unsigned short* __restrict__ K,
                            float* __restrict__ attn) {
    __shared__ float red[8 * 256];
    __shared__ float smat[16][17];

    const int h = blockIdx.x, b = blockIdx.y;
    const int tid = threadIdx.x, wave = tid >> 5, lane = tid & 31;
    const int hi = lane >> 4, lm = lane & 15;

    const unsigned short* qrow = Q + ((size_t)(b * 128 + h * 16 + lm)) * HW;
    const unsigned short* krow = K + ((size_t)(b * 128 + h * 16 + lm)) * HW;

    constexpr int nchunks = (HW + 31) / 32;
    v8f acc = {};
    for (int c = wave; c < nchunks; c += 8) {
        int n0 = c * 32;
        BF16x16 a, bb;
        if (n0 + 32 <= HW) {
            a.f4[0]  = *(const f32x4*)(qrow + n0 + 8 * hi);
            a.f4[1]  = *(const f32x4*)(qrow + n0 + 16 + 8 * hi);
            bb.f4[0] = *(const f32x4*)(krow + n0 + 8 * hi);
            bb.f4[1] = *(const f32x4*)(krow + n0 + 16 + 8 * hi);
        } else {
#pragma unroll
            for (int e = 0; e < 16; ++e) {
                int ka = n0 + e + 8 * ((e >> 3) + hi);
                int kb = n0 + e + 8 * ((e >> 3) + hi);
                unsigned short qv = qrow[imin(ka, HW - 1)];   // unconditional load
                unsigned short kv = krow[imin(kb, HW - 1)];
                a.u[e]  = (ka < HW) ? qv : (unsigned short)0; // cndmask
                bb.u[e] = (kb < HW) ? kv : (unsigned short)0;
            }
        }
        acc = wmma_bf16(a, bb, acc);
    }
#pragma unroll
    for (int r = 0; r < 8; ++r) red[wave * 256 + lane * 8 + r] = acc[r];
    __syncthreads();

    float s = 0.f;
#pragma unroll
    for (int w = 0; w < 8; ++w) s += red[w * 256 + tid];
    int d = (tid & 7) + ((tid >= 128) ? 8 : 0);
    int e = (tid >> 3) & 15;
    smat[d][e] = s * 0.25f;                       // d^-0.5, d=16
    __syncthreads();

    if (tid < 16) {
        int dd = tid;
        float mx = smat[dd][0];
        for (int j = 1; j < 16; ++j) mx = fmaxf(mx, smat[dd][j]);
        float sum = 0.f, ex[16];
        for (int j = 0; j < 16; ++j) { ex[j] = __expf(smat[dd][j] - mx); sum += ex[j]; }
        float inv = 1.f / sum;
        float* ap = attn + (((size_t)b * 8 + h) * 16 + dd) * 16;
        for (int j = 0; j < 16; ++j) ap[j] = ex[j] * inv;
    }
}

// ---------------------------------------------------------------------------
// Effective projection weight: Wp'[b][co][h*16+e] = sum_d Wp[co][h*16+d]*attn
// Folds  conv1x1(wp, attn@v)  into a single per-batch GEMM  Wp' @ V.
// ---------------------------------------------------------------------------
__global__ void wpe_kernel(const float* __restrict__ wp,
                           const float* __restrict__ attn,
                           float* __restrict__ out) {
    const int b = blockIdx.x, tid = threadIdx.x;
    for (int i = tid; i < 128 * 128; i += 256) {
        int co = i >> 7, c = i & 127;
        int h = c >> 4, e = c & 15;
        const float* at = attn + ((size_t)b * 8 + h) * 256 + e;   // [d] stride 16
        const float* w  = wp + (size_t)co * 128 + h * 16;
        float s = 0.f;
#pragma unroll
        for (int d = 0; d < 16; ++d) s += w[d] * at[d * 16];
        out[(size_t)b * 16384 + i] = s;
    }
}

// ---------------------------------------------------------------------------
// conv3x3 'SAME' as implicit-GEMM WMMA.  Input X is NHWC bf16, so each
// B-operand K-chunk is two contiguous b128 loads; 'SAME' padding handled by
// selecting a zeroed page pointer (2 cndmask per tap, not per element).
// Weights staged per-block in LDS as [tap][co][ci] bf16.
// ---------------------------------------------------------------------------
template<int H, int W, int Co>
__global__ void conv3x3_wmma(const unsigned short* __restrict__ X, // [B][H*W][128] bf16
                             const float* __restrict__ Wc,         // [Co][128][3][3]
                             const float* __restrict__ bias,
                             const unsigned short* __restrict__ zeroPage,
                             float* __restrict__ Y) {              // [B][Co][H][W] f32
    __shared__ __align__(16) unsigned short wl[9 * 16 * 128];

    const int b = blockIdx.z;
    const int cobase = blockIdx.y * 16;
    constexpr int nXT = (W + 15) >> 4;
    const int xt = blockIdx.x % nXT;
    const int yb = blockIdx.x / nXT;
    const int tid = threadIdx.x;

    for (int i = tid; i < 9 * 16 * 128; i += 256) {
        int tap = i / 2048, rem = i & 2047;
        int m = rem >> 7, ci = rem & 127;
        int co = cobase + m;
        float wv = (co < Co) ? Wc[((size_t)co * 128 + ci) * 9 + tap] : 0.f;
        wl[i] = f2bf(wv);
    }
    __syncthreads();

    const int wave = tid >> 5, lane = tid & 31;
    const int hi = lane >> 4, lm = lane & 15;
    const int y = yb * 8 + wave;
    const int x0 = xt * 16;
    const int xc = x0 + lm;

    const unsigned short* xb = X + (size_t)b * (H * W) * 128;
    v8f acc = {};

    for (int tap = 0; tap < 9; ++tap) {
        int ky = tap / 3 - 1, kx = tap % 3 - 1;
        int yy = y + ky, xx = xc + kx;
        bool okc = ((unsigned)yy < (unsigned)H) && ((unsigned)xx < (unsigned)W);
        int yyc = imin(imax(yy, 0), H - 1);
        int xxc = imin(imax(xx, 0), W - 1);
        const unsigned short* brow = xb + (yyc * W + xxc) * 128;
        const unsigned short* bsel = okc ? brow : zeroPage;   // pointer cndmask
        const unsigned short* wt = &wl[tap * 2048 + lm * 128];
#pragma unroll
        for (int k0 = 0; k0 < 128; k0 += 32) {
            BF16x16 a, bx;
            a.f4[0] = *(const f32x4*)(wt + k0 + 8 * hi);
            a.f4[1] = *(const f32x4*)(wt + k0 + 16 + 8 * hi);
            const unsigned short* bp = bsel + k0 + 8 * hi;
            bx.f4[0] = *(const f32x4*)(bp);
            bx.f4[1] = *(const f32x4*)(bp + 16);
            acc = wmma_bf16(a, bx, acc);
        }
    }

    if (y < H) {
#pragma unroll
        for (int r = 0; r < 8; ++r) {
            int m = r + 8 * hi;
            int co = cobase + m;
            int xo = x0 + lm;
            if (co < Co && xo < W)
                Y[(((size_t)b * Co + co) * H + y) * W + xo] = acc[r] + bias[co];
        }
    }
}

// ---------------------------------------------------------------------------
// LIF scan: per element, T=8 steps of per-timestep BN + leaky integrate +
// Heaviside(mem > 1) spike with reset-by-subtraction.
// ---------------------------------------------------------------------------
__global__ void lif_kernel(const float* __restrict__ conv,
                           const float* __restrict__ g, const float* __restrict__ bb,
                           const float* __restrict__ m, const float* __restrict__ v,
                           const float* __restrict__ beta,
                           float* __restrict__ out, int Co, int HW) {
    size_t idx = (size_t)blockIdx.x * 256 + threadIdx.x;
    size_t total = (size_t)16 * Co * HW;
    if (idx >= total) return;
    int co = (int)((idx / HW) % Co);
    float x = conv[idx];
    float bc = 1.f / (1.f + __expf(-beta[co]));
    float mem = 0.f;
#pragma unroll
    for (int t = 0; t < 8; ++t) {
        float inv = g[t * Co + co] * rsqrtf(v[t * Co + co] + 1e-5f);
        float xt = x * inv + (bb[t * Co + co] - m[t * Co + co] * inv);
        mem = bc * mem + xt;
        float sp = (mem > 1.f) ? 1.f : 0.f;
        out[(size_t)t * total + idx] = sp;
        mem -= sp;
    }
}

// ---------------------------------------------------------------------------
// Per-scale pipeline (templated so spatial strides become immediates)
// ---------------------------------------------------------------------------
template<int H, int W, int Co>
static void run_scale(const float* featS, const unsigned short* featBf,
                      const float* wq, const float* wk, const float* wv,
                      const float* bng, const float* bnb, const float* bnm, const float* bnv,
                      const float* wproj,
                      const float* cw, const float* cb,
                      const float* mg, const float* mb, const float* mm, const float* mv,
                      const float* beta,
                      unsigned short* Qs, unsigned short* Ks, unsigned short* Vs,
                      unsigned short* Fs, float* Cvs, float* Ats, float* Wpe,
                      const unsigned short* zeroPage,
                      float* outS, hipStream_t stream) {
    constexpr int HW = H * W;
    constexpr int B = 16, C = 128;
    dim3 ggrid((HW + 127) / 128, 8, B);

    gemm_bn_wmma<HW, 1><<<ggrid, 256, 0, stream>>>(featBf, wq, 0ull,
        bng + 0 * C, bnb + 0 * C, bnm + 0 * C, bnv + 0 * C, nullptr, Qs);
    gemm_bn_wmma<HW, 1><<<ggrid, 256, 0, stream>>>(featBf, wk, 0ull,
        bng + 1 * C, bnb + 1 * C, bnm + 1 * C, bnv + 1 * C, nullptr, Ks);
    gemm_bn_wmma<HW, 0><<<ggrid, 256, 0, stream>>>(featBf, wv, 0ull,
        bng + 2 * C, bnb + 2 * C, bnm + 2 * C, bnv + 2 * C, nullptr, Vs);

    attn_kernel<HW><<<dim3(8, B), 256, 0, stream>>>(Qs, Ks, Ats);
    wpe_kernel<<<B, 256, 0, stream>>>(wproj, Ats, Wpe);

    gemm_bn_wmma<HW, 2><<<ggrid, 256, 0, stream>>>(Vs, Wpe, 16384ull,
        nullptr, nullptr, nullptr, nullptr, featS, Fs);

    constexpr int nXT = (W + 15) / 16, nYB = (H + 7) / 8, coT = (Co + 15) / 16;
    conv3x3_wmma<H, W, Co><<<dim3(nXT * nYB, coT, B), 256, 0, stream>>>(
        Fs, cw, cb, zeroPage, Cvs);

    size_t total = (size_t)B * Co * HW;
    lif_kernel<<<(unsigned)((total + 255) / 256), 256, 0, stream>>>(
        Cvs, mg, mb, mm, mv, beta, outS, Co, HW);
}

// ---------------------------------------------------------------------------
// Host orchestration
// ---------------------------------------------------------------------------
extern "C" void kernel_launch(void* const* d_in, const int* in_sizes, int n_in,
                              void* d_out, int out_size, void* d_ws, size_t ws_size,
                              hipStream_t stream) {
    (void)in_sizes; (void)n_in; (void)out_size; (void)ws_size;

    const float* feat   = (const float*)d_in[0];
    const float* w_qkv  = (const float*)d_in[1];   // [3][3][128][128]
    const float* w_proj = (const float*)d_in[2];   // [3][128][128]
    const float* bn_g   = (const float*)d_in[3];   // [3][3][128]
    const float* bn_b   = (const float*)d_in[4];
    const float* bn_m   = (const float*)d_in[5];
    const float* bn_v   = (const float*)d_in[6];
    // d_in[7] = slope: forward Heaviside is slope-independent
    const float* conv_w[3] = { (const float*)d_in[8],  (const float*)d_in[15], (const float*)d_in[22] };
    const float* conv_b[3] = { (const float*)d_in[9],  (const float*)d_in[16], (const float*)d_in[23] };
    const float* mg[3]     = { (const float*)d_in[10], (const float*)d_in[17], (const float*)d_in[24] };
    const float* mb[3]     = { (const float*)d_in[11], (const float*)d_in[18], (const float*)d_in[25] };
    const float* mm[3]     = { (const float*)d_in[12], (const float*)d_in[19], (const float*)d_in[26] };
    const float* mv[3]     = { (const float*)d_in[13], (const float*)d_in[20], (const float*)d_in[27] };
    const float* beta[3]   = { (const float*)d_in[14], (const float*)d_in[21], (const float*)d_in[28] };
    float* out = (float*)d_out;

    const int B = 16, C = 128;
    const int HWs[3] = { 6400, 1600, 400 };
    const int Cos[3] = { 12, 18, 24 };

    // ---- workspace allocation (256B aligned).
    // F_s (bf16 NHWC) aliases Q_s|K_s; conv-out (f32) aliases V_s.
    char* ws = (char*)d_ws;
    size_t off = 0;
    auto alloc = [&](size_t bytes) { size_t o = off; off += (bytes + 255) & ~(size_t)255; return o; };

    size_t o_zero  = alloc(256);
    size_t o_feat2 = alloc((size_t)B * C * 1600 * 4);
    size_t o_feat4 = alloc((size_t)B * C * 400 * 4);
    size_t o_fbf[3], o_qk[3], o_v[3], o_attn[3], o_wpe[3];
    for (int s = 0; s < 3; ++s) {
        size_t HW = HWs[s];
        o_fbf[s]  = alloc((size_t)B * HW * 128 * 2);     // feat bf16 NHWC
        o_qk[s]   = alloc(2 * (size_t)B * C * HW * 2);   // Q + K (bf16 NCHW); F aliases
        o_v[s]    = alloc((size_t)B * HW * 128 * 2);     // V (bf16 NHWC); conv-out aliases
        o_attn[s] = alloc((size_t)B * 8 * 16 * 16 * 4);
        o_wpe[s]  = alloc((size_t)B * 128 * 128 * 4);
    }

    unsigned short* zeroPage = (unsigned short*)(ws + o_zero);
    float* feat2 = (float*)(ws + o_feat2);
    float* feat4 = (float*)(ws + o_feat4);

    zero_kernel<<<1, 128, 0, stream>>>(zeroPage);
    pool_kernel<2, 40, 40><<<(B * C * 1600 + 255) / 256, 256, 0, stream>>>(feat, feat2);
    pool_kernel<4, 20, 20><<<(B * C * 400  + 255) / 256, 256, 0, stream>>>(feat, feat4);

    const float* feats[3] = { feat, feat2, feat4 };
    unsigned short* fbf[3];
    for (int s = 0; s < 3; ++s) fbf[s] = (unsigned short*)(ws + o_fbf[s]);

    castT_kernel<6400><<<dim3(6400 / 16, B), 256, 0, stream>>>(feats[0], fbf[0]);
    castT_kernel<1600><<<dim3(1600 / 16, B), 256, 0, stream>>>(feats[1], fbf[1]);
    castT_kernel< 400><<<dim3( 400 / 16, B), 256, 0, stream>>>(feats[2], fbf[2]);

    size_t obase[3] = { 0,
                        (size_t)8 * B * Cos[0] * HWs[0],
                        (size_t)8 * B * Cos[0] * HWs[0] + (size_t)8 * B * Cos[1] * HWs[1] };

    for (int s = 0; s < 3; ++s) {
        unsigned short* Qs = (unsigned short*)(ws + o_qk[s]);
        unsigned short* Ks = Qs + (size_t)B * C * HWs[s];
        unsigned short* Vs = (unsigned short*)(ws + o_v[s]);
        unsigned short* Fs = (unsigned short*)(ws + o_qk[s]);
        float* Cvs = (float*)(ws + o_v[s]);
        float* Ats = (float*)(ws + o_attn[s]);
        float* Wpe = (float*)(ws + o_wpe[s]);
        const float* wq = w_qkv + ((size_t)s * 3 + 0) * 16384;
        const float* wk = w_qkv + ((size_t)s * 3 + 1) * 16384;
        const float* wv = w_qkv + ((size_t)s * 3 + 2) * 16384;
        const float* bg = bn_g + (size_t)s * 3 * C;
        const float* bb = bn_b + (size_t)s * 3 * C;
        const float* bm = bn_m + (size_t)s * 3 * C;
        const float* bv = bn_v + (size_t)s * 3 * C;
        const float* wp = w_proj + (size_t)s * 16384;

        if (s == 0)
            run_scale<80, 80, 12>(feats[s], fbf[s], wq, wk, wv, bg, bb, bm, bv, wp,
                conv_w[s], conv_b[s], mg[s], mb[s], mm[s], mv[s], beta[s],
                Qs, Ks, Vs, Fs, Cvs, Ats, Wpe, zeroPage, out + obase[s], stream);
        else if (s == 1)
            run_scale<40, 40, 18>(feats[s], fbf[s], wq, wk, wv, bg, bb, bm, bv, wp,
                conv_w[s], conv_b[s], mg[s], mb[s], mm[s], mv[s], beta[s],
                Qs, Ks, Vs, Fs, Cvs, Ats, Wpe, zeroPage, out + obase[s], stream);
        else
            run_scale<20, 20, 24>(feats[s], fbf[s], wq, wk, wv, bg, bb, bm, bv, wp,
                conv_w[s], conv_b[s], mg[s], mb[s], mm[s], mv[s], beta[s],
                Qs, Ks, Vs, Fs, Cvs, Ats, Wpe, zeroPage, out + obase[s], stream);
    }
}
